// BidirLinearAttentionLayer_82360292868683
// MI455X (gfx1250) — matrix operand order repaired and checked by reading, not compiled, not run on hardware
//
#include <hip/hip_runtime.h>
#include <math.h>

// ---------------------------------------------------------------------------
// Types for CDNA5 WMMA (wave32, 16x16x32 bf16 -> f32)
// ---------------------------------------------------------------------------
typedef __attribute__((ext_vector_type(16))) __bf16        v16bf;
typedef __attribute__((ext_vector_type(8)))  float         v8f;
typedef __attribute__((ext_vector_type(4)))  unsigned int  u32x4;

union FragAB { v16bf v; u32x4 q[2]; };

static constexpr int Bc = 2, Tc = 2048, Dc = 512, Hc = 8, HDc = 64, Fc = 2048;

// ---------------------------------------------------------------------------
// fp32 -> bf16 elementwise convert (weights)
// ---------------------------------------------------------------------------
__global__ void cvt_bf16_kernel(const float* __restrict__ in, __bf16* __restrict__ out, int n) {
  int i = blockIdx.x * blockDim.x + threadIdx.x;
  if (i < n) out[i] = (__bf16)in[i];
}

// ---------------------------------------------------------------------------
// LayerNorm over D=512 per row, write bf16 (GEMM input)
// ---------------------------------------------------------------------------
__global__ __launch_bounds__(128)
void ln_bf16_kernel(const float* __restrict__ x, const float* __restrict__ g,
                    const float* __restrict__ b, __bf16* __restrict__ out) {
  const int Dn = Dc;
  int row = blockIdx.x, tid = threadIdx.x;
  const float* xr = x + (size_t)row * Dn;
  float s = 0.f, s2 = 0.f;
  for (int i = tid; i < Dn; i += blockDim.x) { float v = xr[i]; s += v; s2 += v * v; }
  __shared__ float r1[128], r2[128];
  r1[tid] = s; r2[tid] = s2; __syncthreads();
  for (int off = 64; off > 0; off >>= 1) {
    if (tid < off) { r1[tid] += r1[tid + off]; r2[tid] += r2[tid + off]; }
    __syncthreads();
  }
  float mean = r1[0] / Dn;
  float var  = r2[0] / Dn - mean * mean;
  float rstd = rsqrtf(var + 1e-5f);
  __bf16* orow = out + (size_t)row * Dn;
  for (int i = tid; i < Dn; i += blockDim.x)
    orow[i] = (__bf16)(((xr[i] - mean) * rstd) * g[i] + b[i]);
}

// ---------------------------------------------------------------------------
// Generic bf16 WMMA GEMM: out = A[M,K] * W[N,K]^T  (both row-major, K inner)
// Block: 256 threads (8 waves) -> 128x128 tile; wave -> 64(M)x32(N) via 4x2 WMMA.
// Fragment layouts per CDNA5 ISA 7.12.2:
//   A 16x32 bf16 : lane(0-15)=row m, K 0..7 & 16..23 ; lane(16-31)=row m, K 8..15 & 24..31
//   B 32x16 bf16 : lane(0-15)=col n, K 0..7 & 16..23 ; lane(16-31)=col n, K 8..15 & 24..31
//   C/D 16x16 f32: vgpr r, lane l -> m = r + 8*(l>>4), n = l&15
// ---------------------------------------------------------------------------
enum { EPI_Q = 0, EPI_K = 1, EPI_V = 2, EPI_O = 3, EPI_F1 = 4, EPI_F2 = 5 };

template<int MODE>
__global__ __launch_bounds__(256)
void gemm_bf16_kernel(const __bf16* __restrict__ A, const __bf16* __restrict__ W,
                      int M, int N, int K,
                      const float* __restrict__ bias, const float* __restrict__ resid,
                      const unsigned char* __restrict__ mask,
                      float* __restrict__ outf, __bf16* __restrict__ outb)
{
  const int lane = threadIdx.x & 31;
  const int wid  = threadIdx.x >> 5;
  const int wm = wid >> 2, wn = wid & 3;          // 2 (M) x 4 (N) wave grid
  const int blockM = blockIdx.y * 128;
  const int blockN = blockIdx.x * 128;
  const int khalf = lane >> 4, l16 = lane & 15;

  v8f acc[4][2];
  #pragma unroll
  for (int i = 0; i < 4; ++i)
    #pragma unroll
    for (int j = 0; j < 2; ++j)
      acc[i][j] = (v8f){0.f,0.f,0.f,0.f,0.f,0.f,0.f,0.f};

  const __bf16* Arow[4];
  const __bf16* Wrow[2];
  #pragma unroll
  for (int i = 0; i < 4; ++i)
    Arow[i] = A + (size_t)(blockM + wm * 64 + i * 16 + l16) * K;
  #pragma unroll
  for (int j = 0; j < 2; ++j)
    Wrow[j] = W + (size_t)(blockN + wn * 32 + j * 16 + l16) * K;

  for (int k0 = 0; k0 < K; k0 += 32) {
    FragAB af[4], bf[2];
    #pragma unroll
    for (int i = 0; i < 4; ++i) {
      const __bf16* p = Arow[i] + k0 + khalf * 8;
      af[i].q[0] = *(const u32x4*)(p);
      af[i].q[1] = *(const u32x4*)(p + 16);
    }
    #pragma unroll
    for (int j = 0; j < 2; ++j) {
      const __bf16* p = Wrow[j] + k0 + khalf * 8;
      bf[j].q[0] = *(const u32x4*)(p);
      bf[j].q[1] = *(const u32x4*)(p + 16);
    }
    if (k0 + 32 < K) {                  // near-cache hint for next K tile
      __builtin_prefetch(Arow[0] + k0 + 32, 0, 3);
      __builtin_prefetch(Wrow[0] + k0 + 32, 0, 3);
    }
    #pragma unroll
    for (int i = 0; i < 4; ++i) {
      #pragma unroll
      for (int j = 0; j < 2; ++j)
        acc[i][j] = __builtin_amdgcn_wmma_f32_16x16x32_bf16(
            false, af[i].v, false, bf[j].v, (short)0, acc[i][j], false, false);
    }
  }

  #pragma unroll
  for (int i = 0; i < 4; ++i) {
    #pragma unroll
    for (int j = 0; j < 2; ++j) {
      v8f a = acc[i][j];
      #pragma unroll
      for (int r = 0; r < 8; ++r) {
        int m = blockM + wm * 64 + i * 16 + khalf * 8 + r;
        int n = blockN + wn * 32 + j * 16 + l16;
        float v = a[r];
        if constexpr (MODE == EPI_Q || MODE == EPI_K || MODE == EPI_V) {
          int bb = m >> 11;              // T = 2048
          int tt = m & (Tc - 1);
          int hh = n >> 6;               // hd = 64
          int ee = n & (HDc - 1);
          if constexpr (MODE == EPI_Q || MODE == EPI_K)
            v = (v > 0.f) ? (v + 1.f) : __expf(v);   // elu(x)+1
          if constexpr (MODE == EPI_K || MODE == EPI_V)
            v = mask[(size_t)bb * Tc + tt] ? 0.f : v;
          size_t dst = (((size_t)(bb * Hc + hh) * Tc) + tt) * HDc + ee;
          outf[dst] = v;
        } else if constexpr (MODE == EPI_O) {
          size_t idx = (size_t)m * N + n;
          outf[idx] = v + bias[n] + resid[idx];
        } else if constexpr (MODE == EPI_F1) {
          float u  = v + bias[n];
          float gl = 0.5f * u * (1.f + erff(u * 0.70710678118654752f));  // exact gelu
          outb[(size_t)m * N + n] = (__bf16)gl;
        } else {  // EPI_F2
          size_t idx = (size_t)m * N + n;
          outf[idx] = v + bias[n] + resid[idx];
        }
      }
    }
  }
}

// ---------------------------------------------------------------------------
// Async global->LDS staging helpers (CDNA5 GLOBAL_LOAD_ASYNC_TO_LDS_B32,
// tracked by ASYNCcnt).  LDS destination address = low 32 bits of the generic
// pointer.
// ---------------------------------------------------------------------------
__device__ __forceinline__ void async_g2l_b32(void* lds_dst, const void* gsrc) {
  unsigned dst = (unsigned)(unsigned long long)lds_dst;
  asm volatile("global_load_async_to_lds_b32 %0, %1, off"
               :: "v"(dst), "v"(gsrc) : "memory");
}
__device__ __forceinline__ void wait_asynccnt0() {
#if __has_builtin(__builtin_amdgcn_s_wait_asynccnt)
  __builtin_amdgcn_s_wait_asynccnt(0);
#else
  asm volatile("s_wait_asynccnt 0x0" ::: "memory");
#endif
}

// ---------------------------------------------------------------------------
// Linear-attention recurrence. One 64-thread block per (b,h); thread e owns
// S[:,e] (64 regs) and a replicated z. 32-step q/k/v chunks staged in LDS
// via async global->LDS copies.
// ---------------------------------------------------------------------------
template<bool REV>
__global__ __launch_bounds__(64)
void recur_kernel(const float* __restrict__ Q, const float* __restrict__ Kf,
                  const float* __restrict__ V, const float* __restrict__ decay_logit,
                  float* __restrict__ Y, float* __restrict__ C)
{
  const int bh = blockIdx.x;                 // 0..15
  const int h  = bh & (Hc - 1);
  const int e  = threadIdx.x;                // 0..63
  const float lam = 1.f / (1.f + __expf(-decay_logit[h]));
  const size_t base = (size_t)bh * Tc * HDc;

  float S[HDc], z[HDc];
  #pragma unroll
  for (int i = 0; i < HDc; ++i) { S[i] = 0.f; z[i] = 0.f; }

  constexpr int CH = 32;
  __shared__ float lq[CH][HDc], lk[CH][HDc], lv[CH][HDc];

  #pragma unroll 1
  for (int c0 = 0; c0 < Tc; c0 += CH) {
    __syncthreads();
    #pragma unroll 1
    for (int tt = 0; tt < CH; ++tt) {
      int t = REV ? (Tc - 1 - (c0 + tt)) : (c0 + tt);
      size_t idx = base + (size_t)t * HDc + e;
      async_g2l_b32(&lq[tt][e], Q + idx);
      async_g2l_b32(&lk[tt][e], Kf + idx);
      async_g2l_b32(&lv[tt][e], V + idx);
    }
    wait_asynccnt0();
    __syncthreads();
    #pragma unroll 1
    for (int tt = 0; tt < CH; ++tt) {
      float ve = lv[tt][e];
      float y = 0.f, c = 0.f, qk = 0.f;
      #pragma unroll
      for (int i = 0; i < HDc; ++i) {
        float qi = lq[tt][i], ki = lk[tt][i];
        S[i] = lam * S[i] + ki * ve;
        y += qi * S[i];
        z[i] = lam * z[i] + ki;      // replicated across threads (identical)
        c += qi * z[i];
        qk += qi * ki;
      }
      int t = REV ? (Tc - 1 - (c0 + tt)) : (c0 + tt);
      Y[base + (size_t)t * HDc + e] = y - 0.5f * qk * ve;
      if (e == 0) C[(size_t)bh * Tc + t] = c - 0.5f * qk;
    }
  }
}

// ---------------------------------------------------------------------------
// attn = (Yf+Yb) / max(Cf+Cb, 1e-6), scattered back to [B,T,D] as bf16
// ---------------------------------------------------------------------------
__global__ void combine_kernel(const float* __restrict__ Yf, const float* __restrict__ Yb,
                               const float* __restrict__ Cf, const float* __restrict__ Cb,
                               __bf16* __restrict__ attn)
{
  int gid = blockIdx.x * blockDim.x + threadIdx.x;   // over B*H*T*hd
  int e  = gid & (HDc - 1);
  int t  = (gid >> 6) & (Tc - 1);
  int bh = gid >> 17;                                 // T*hd = 2^17
  int b  = bh >> 3, h = bh & 7;
  float y = Yf[gid] + Yb[gid];
  float c = Cf[(bh << 11) + t] + Cb[(bh << 11) + t];
  c = fmaxf(c, 1e-6f);
  attn[((size_t)(b * Tc + t) << 9) + (h << 6) + e] = (__bf16)(y / c);
}

// ---------------------------------------------------------------------------
// Orchestration
// ---------------------------------------------------------------------------
extern "C" void kernel_launch(void* const* d_in, const int* in_sizes, int n_in,
                              void* d_out, int out_size, void* d_ws, size_t ws_size,
                              hipStream_t stream)
{
  (void)in_sizes; (void)n_in; (void)out_size; (void)ws_size;

  const float*         x    = (const float*)d_in[0];
  const unsigned char* mask = (const unsigned char*)d_in[1];
  const float* Wq  = (const float*)d_in[2];
  const float* Wk  = (const float*)d_in[3];
  const float* Wv  = (const float*)d_in[4];
  const float* Wo  = (const float*)d_in[5];
  const float* bo  = (const float*)d_in[6];
  const float* g1  = (const float*)d_in[7];
  const float* b1  = (const float*)d_in[8];
  const float* g2  = (const float*)d_in[9];
  const float* b2  = (const float*)d_in[10];
  const float* W1  = (const float*)d_in[11];
  const float* bf1 = (const float*)d_in[12];
  const float* W2  = (const float*)d_in[13];
  const float* bf2 = (const float*)d_in[14];
  const float* dl  = (const float*)d_in[15];

  char* ws = (char*)d_ws;
  size_t off = 0;
  auto alloc = [&](size_t bytes) { size_t r = off; off += (bytes + 255) & ~(size_t)255; return r; };

  const size_t M = (size_t)Bc * Tc;   // 4096
  size_t o_xn = alloc(M * Dc * 2);
  size_t o_wq = alloc((size_t)Dc * Dc * 2);
  size_t o_wk = alloc((size_t)Dc * Dc * 2);
  size_t o_wv = alloc((size_t)Dc * Dc * 2);
  size_t o_wo = alloc((size_t)Dc * Dc * 2);
  size_t o_w1 = alloc((size_t)Fc * Dc * 2);
  size_t o_w2 = alloc((size_t)Dc * Fc * 2);
  size_t o_Q  = alloc(M * Dc * 4);
  size_t o_K  = alloc(M * Dc * 4);
  size_t o_V  = alloc(M * Dc * 4);
  size_t o_Yf = alloc(M * Dc * 4);
  size_t o_Yb = alloc(M * Dc * 4);        // contiguous with o_Yf
  size_t o_Cf = alloc((size_t)Bc * Hc * Tc * 4);
  size_t o_Cb = alloc((size_t)Bc * Hc * Tc * 4);
  size_t o_at = alloc(M * Dc * 2);
  // buffer reuse after their producers are dead:
  size_t o_x2 = o_Q;    // fp32 [M,D]  (Q dead after recurrences)
  size_t o_h  = o_K;    // bf16 [M,D]  (K dead)
  size_t o_ff = o_Yf;   // bf16 [M,F] spans Yf+Yb (16 MB, dead after combine)

  __bf16* xn_bf = (__bf16*)(ws + o_xn);
  __bf16* wq_bf = (__bf16*)(ws + o_wq);
  __bf16* wk_bf = (__bf16*)(ws + o_wk);
  __bf16* wv_bf = (__bf16*)(ws + o_wv);
  __bf16* wo_bf = (__bf16*)(ws + o_wo);
  __bf16* w1_bf = (__bf16*)(ws + o_w1);
  __bf16* w2_bf = (__bf16*)(ws + o_w2);
  float*  Qh    = (float*)(ws + o_Q);
  float*  Kh    = (float*)(ws + o_K);
  float*  Vh    = (float*)(ws + o_V);
  float*  Yf    = (float*)(ws + o_Yf);
  float*  Yb    = (float*)(ws + o_Yb);
  float*  Cf    = (float*)(ws + o_Cf);
  float*  Cb    = (float*)(ws + o_Cb);
  __bf16* at_bf = (__bf16*)(ws + o_at);
  float*  x2    = (float*)(ws + o_x2);
  __bf16* h_bf  = (__bf16*)(ws + o_h);
  __bf16* ff_bf = (__bf16*)(ws + o_ff);

  const int nDD = Dc * Dc, nFD = Fc * Dc;
  cvt_bf16_kernel<<<(nDD + 255) / 256, 256, 0, stream>>>(Wq, wq_bf, nDD);
  cvt_bf16_kernel<<<(nDD + 255) / 256, 256, 0, stream>>>(Wk, wk_bf, nDD);
  cvt_bf16_kernel<<<(nDD + 255) / 256, 256, 0, stream>>>(Wv, wv_bf, nDD);
  cvt_bf16_kernel<<<(nDD + 255) / 256, 256, 0, stream>>>(Wo, wo_bf, nDD);
  cvt_bf16_kernel<<<(nFD + 255) / 256, 256, 0, stream>>>(W1, w1_bf, nFD);
  cvt_bf16_kernel<<<(nFD + 255) / 256, 256, 0, stream>>>(W2, w2_bf, nFD);

  ln_bf16_kernel<<<(unsigned)M, 128, 0, stream>>>(x, g1, b1, xn_bf);

  dim3 blk(256);
  dim3 grdD(Dc / 128, (unsigned)(M / 128));   // (4, 32)
  gemm_bf16_kernel<EPI_Q><<<grdD, blk, 0, stream>>>(xn_bf, wq_bf, (int)M, Dc, Dc, nullptr, nullptr, nullptr, Qh, nullptr);
  gemm_bf16_kernel<EPI_K><<<grdD, blk, 0, stream>>>(xn_bf, wk_bf, (int)M, Dc, Dc, nullptr, nullptr, mask, Kh, nullptr);
  gemm_bf16_kernel<EPI_V><<<grdD, blk, 0, stream>>>(xn_bf, wv_bf, (int)M, Dc, Dc, nullptr, nullptr, mask, Vh, nullptr);

  recur_kernel<false><<<Bc * Hc, 64, 0, stream>>>(Qh, Kh, Vh, dl, Yf, Cf);
  recur_kernel<true ><<<Bc * Hc, 64, 0, stream>>>(Qh, Kh, Vh, dl, Yb, Cb);

  combine_kernel<<<(unsigned)(M * Dc / 256), 256, 0, stream>>>(Yf, Yb, Cf, Cb, at_bf);

  gemm_bf16_kernel<EPI_O><<<grdD, blk, 0, stream>>>(at_bf, wo_bf, (int)M, Dc, Dc, bo, x, nullptr, x2, nullptr);

  ln_bf16_kernel<<<(unsigned)M, 128, 0, stream>>>(x2, g2, b2, h_bf);

  dim3 grdF(Fc / 128, (unsigned)(M / 128));   // (16, 32)
  gemm_bf16_kernel<EPI_F1><<<grdF, blk, 0, stream>>>(h_bf, w1_bf, (int)M, Fc, Dc, bf1, nullptr, nullptr, nullptr, ff_bf);
  gemm_bf16_kernel<EPI_F2><<<grdD, blk, 0, stream>>>(ff_bf, w2_bf, (int)M, Dc, Fc, bf2, x2, nullptr, (float*)d_out, nullptr);
}
